// Attention_63367947485272
// MI455X (gfx1250) — compile-verified, matrix operand order checked by reading, not run on hardware
//
#include <hip/hip_runtime.h>
#include <hip/hip_bf16.h>
#include <math.h>

// ---------------------------------------------------------------------------
// Problem constants (B,N,C,H,D from the reference)
// B=2, N=2048, C=1024, H=16, D=64, M = B*N = 4096
// ---------------------------------------------------------------------------
#define NPOS 2048
#define CDIM 1024
#define HDIM 16
#define DDIM 64
#define MROWS 4096
// softmax scale (D^-0.5) folded with log2(e) so we can use exp2
#define QSCALE 0.18033688011112042f  // 0.125 * 1.4426950408889634

typedef __attribute__((ext_vector_type(16))) __bf16 v16bf;
typedef __attribute__((ext_vector_type(8)))  float  v8f;

union Frag {
  v16bf v;
  uint4 q[2];
  unsigned short s[16];
};

__device__ __forceinline__ unsigned short f2bf(float f) {
  unsigned int u = __float_as_uint(f);
  return (unsigned short)((u + 0x7FFFu + ((u >> 16) & 1u)) >> 16);
}

// ---------------------------------------------------------------------------
// Async global->LDS 128-bit copy (CDNA5 GLOBAL_LOAD_ASYNC_TO_LDS_B128,
// tracked by ASYNCcnt). Builtin signature (from hipcc diagnostic):
//   (int4-vec __device__*  global_src, int4-vec __shared__* lds_dst,
//    imm offset, imm cpol)
// ---------------------------------------------------------------------------
typedef int v4i __attribute__((vector_size(16)));
typedef __attribute__((address_space(1))) v4i v4i_g;
typedef __attribute__((address_space(3))) v4i v4i_l;

__device__ __forceinline__ void async_copy_b128(const unsigned short* gsrc,
                                                unsigned short* ldst) {
#if __has_builtin(__builtin_amdgcn_global_load_async_to_lds_b128)
  __builtin_amdgcn_global_load_async_to_lds_b128(
      (v4i_g*)gsrc, (v4i_l*)ldst, /*offset=*/0, /*cpol=*/0);
#else
  // vdst = LDS byte address (low 32 bits of generic shared pointer),
  // vaddr = 64-bit global address
  asm volatile("global_load_async_to_lds_b128 %0, %1, off"
               :
               : "v"((unsigned)(unsigned long long)ldst), "v"(gsrc)
               : "memory");
#endif
}

__device__ __forceinline__ void wait_async() {
#if __has_builtin(__builtin_amdgcn_s_wait_asynccnt)
  __builtin_amdgcn_s_wait_asynccnt(0);
#else
  asm volatile("s_wait_asynccnt 0x0" ::: "memory");
#endif
}

// Load a 16x32 bf16 WMMA operand fragment from a row-major LDS tile.
// Per the CDNA5 16-bit A/B layout: lane holds row (ln), element j:
//   lanes 0-15 : k = {0..7, 16..23}, lanes 16-31 : k = {8..15, 24..31}
// -> two contiguous 16-byte LDS loads at kbase+8*hl and kbase+16+8*hl.
__device__ __forceinline__ void ld_frag(Frag& f, const unsigned short* lds,
                                        int row, int stride, int kbase, int hl) {
  const unsigned short* p = lds + row * stride + kbase + 8 * hl;
  f.q[0] = *(const uint4*)(p);
  f.q[1] = *(const uint4*)(p + 16);
}

__device__ __forceinline__ v8f wmma_bf16(const Frag& a, const Frag& b, v8f c) {
  return __builtin_amdgcn_wmma_f32_16x16x32_bf16(
      /*neg_a=*/false, a.v, /*neg_b=*/false, b.v,
      /*c_mod=*/(short)0, c, /*reuse_a=*/false, /*reuse_b=*/false);
}

// ---------------------------------------------------------------------------
// fp32 -> bf16 cast (grid-stride)
// ---------------------------------------------------------------------------
__global__ void cast_f32_bf16(const float* __restrict__ in,
                              unsigned short* __restrict__ out, int n) {
  int i = blockIdx.x * blockDim.x + threadIdx.x;
  int stride = gridDim.x * blockDim.x;
  for (; i < n; i += stride) out[i] = f2bf(in[i]);
}

// ---------------------------------------------------------------------------
// bf16 WMMA GEMM: C[M,Ng] = A[M,K] * B[K,Ng]
// Block: 256 thr = 8 waves, tile 64(M) x 256(N); wave tile 32x64 (2x4 accs).
// A tile staged with async global->LDS; B tile transposed through registers.
// MODE 0: f32 out + bias (projection -> d_out)
// MODE 1: scatter into per-head q/k/v bf16 buffers, q pre-scaled by QSCALE
// ---------------------------------------------------------------------------
template <int MODE>
__global__ __launch_bounds__(256) void gemm_wmma(
    const unsigned short* __restrict__ A, const unsigned short* __restrict__ Bm,
    int Ng, int K,
    float* __restrict__ Cout, const float* __restrict__ bias,
    unsigned short* __restrict__ qo, unsigned short* __restrict__ ko,
    unsigned short* __restrict__ vo) {
  __shared__ __align__(16) unsigned short As[64 * 40];   // 64 rows x 32k (+8 pad)
  __shared__ __align__(16) unsigned short Bt[256 * 40];  // 256 n-rows x 32k (+8 pad)

  const int tid = threadIdx.x, lane = tid & 31, wid = tid >> 5;
  const int ln = lane & 15, hl = lane >> 4;
  const int wm = wid & 1, wn = wid >> 1;
  const int m0 = blockIdx.y * 64, n0 = blockIdx.x * 256;

  v8f acc[2][4];
#pragma unroll
  for (int mt = 0; mt < 2; mt++)
#pragma unroll
    for (int nt = 0; nt < 4; nt++) acc[mt][nt] = (v8f){0, 0, 0, 0, 0, 0, 0, 0};

  for (int k0 = 0; k0 < K; k0 += 32) {
    // stage A tile 64x32 row-major: async 128b global->LDS per thread
    {
      int row = tid >> 2, seg = tid & 3;
      if (k0 + 32 < K)
        __builtin_prefetch(A + (size_t)(m0 + row) * K + k0 + 32, 0, 1);
      async_copy_b128(A + (size_t)(m0 + row) * K + k0 + seg * 8,
                      As + row * 40 + seg * 8);
    }
    // stage B tile transposed: Bt[n][k], coalesced global reads + LDS scatter
    {
      int kr = tid >> 3, nseg = (tid & 7) * 32;
      union { uint4 q[4]; unsigned short s[32]; } tmp;
      const uint4* src = (const uint4*)(Bm + (size_t)(k0 + kr) * Ng + n0 + nseg);
      tmp.q[0] = src[0]; tmp.q[1] = src[1]; tmp.q[2] = src[2]; tmp.q[3] = src[3];
#pragma unroll
      for (int j = 0; j < 32; j++) Bt[(nseg + j) * 40 + kr] = tmp.s[j];
    }
    wait_async();
    __syncthreads();

    Frag bf[4];
#pragma unroll
    for (int nt = 0; nt < 4; nt++)
      ld_frag(bf[nt], Bt, wn * 64 + nt * 16 + ln, 40, 0, hl);
#pragma unroll
    for (int mt = 0; mt < 2; mt++) {
      Frag af;
      ld_frag(af, As, wm * 32 + mt * 16 + ln, 40, 0, hl);
#pragma unroll
      for (int nt = 0; nt < 4; nt++) acc[mt][nt] = wmma_bf16(af, bf[nt], acc[mt][nt]);
    }
    __syncthreads();
  }

  // epilogue (C-layout: lane half hl gives rows hl*8+r, column = ln)
#pragma unroll
  for (int mt = 0; mt < 2; mt++)
#pragma unroll
    for (int nt = 0; nt < 4; nt++)
#pragma unroll
      for (int r = 0; r < 8; r++) {
        int rg = m0 + wm * 32 + mt * 16 + hl * 8 + r;
        int cg = n0 + wn * 64 + nt * 16 + ln;
        float v = acc[mt][nt][r];
        if (MODE == 0) {
          Cout[(size_t)rg * Ng + cg] = v + bias[cg];
        } else {
          // cg in [0,3C): three = cg/1024, head = (cg%1024)/64, d = cg%64
          int three = cg >> 10, rem = cg & 1023;
          int hh = rem >> 6, d = rem & 63;
          int bb = rg >> 11, np = rg & 2047;
          size_t off = (((size_t)bb * HDIM + hh) * NPOS + np) * DDIM + d;
          if (three == 0) v *= QSCALE;  // fold softmax scale * log2(e) into Q
          unsigned short* dst = (three == 0) ? qo : ((three == 1) ? ko : vo);
          dst[off] = f2bf(v);
        }
      }
}

// ---------------------------------------------------------------------------
// Flash attention: grid (N/128, B*H), 256 thr = 8 waves x 16 queries.
// Scores computed transposed (S^T = K_tile x Q^T) so softmax is lane-local
// (+1 shfl_xor) and exp(P) lands directly in the A-operand layout for P x V.
// Q and K tiles staged with async global->LDS; V transposed through registers.
// ---------------------------------------------------------------------------
__global__ __launch_bounds__(256) void attn_kernel(
    const unsigned short* __restrict__ Qg, const unsigned short* __restrict__ Kg,
    const unsigned short* __restrict__ Vg, unsigned short* __restrict__ Og) {
  __shared__ __align__(16) unsigned short Qs[128 * 72];  // [query][d]
  __shared__ __align__(16) unsigned short Ks[64 * 72];   // [key][d]
  __shared__ __align__(16) unsigned short Vt[64 * 72];   // [d][key] (transposed)

  const int tid = threadIdx.x, lane = tid & 31, wave = tid >> 5;
  const int ln = lane & 15, hl = lane >> 4;
  const int bh = blockIdx.y;  // b*H + h
  const int b = bh >> 4, h = bh & 15;
  const int qbase = blockIdx.x * 128;
  const size_t headOff = (size_t)bh * NPOS * DDIM;

  // stage Q tile [128][64] once -- 4 async 128b copies per thread
  {
    int row = tid >> 1, seg = tid & 1;
    const unsigned short* src = Qg + headOff + (size_t)(qbase + row) * DDIM + seg * 32;
    unsigned short* dst = Qs + row * 72 + seg * 32;
#pragma unroll
    for (int c = 0; c < 4; c++) async_copy_b128(src + c * 8, dst + c * 8);
  }

  v8f accO[4];
#pragma unroll
  for (int dt = 0; dt < 4; dt++) accO[dt] = (v8f){0, 0, 0, 0, 0, 0, 0, 0};
  float m_run = -1e30f, l_run = 0.0f;

  for (int kt = 0; kt < NPOS; kt += 64) {
    __syncthreads();  // previous tile consumed before restaging
    // stage K tile [64][64] row-major: 2 async 128b copies per thread
    {
      int row = tid >> 2, seg = tid & 3;
      const unsigned short* src = Kg + headOff + (size_t)(kt + row) * DDIM + seg * 16;
      unsigned short* dst = Ks + row * 72 + seg * 16;
      async_copy_b128(src, dst);
      async_copy_b128(src + 8, dst + 8);
    }
    // stage V transposed: Vt[d][key] (coalesced read, LDS scatter)
    {
      int key = tid >> 2, dseg = (tid & 3) * 16;
      union { uint4 q[2]; unsigned short s[16]; } tmp;
      const uint4* src = (const uint4*)(Vg + headOff + (size_t)(kt + key) * DDIM + dseg);
      tmp.q[0] = src[0]; tmp.q[1] = src[1];
#pragma unroll
      for (int j = 0; j < 16; j++) Vt[(dseg + j) * 72 + key] = tmp.s[j];
    }
    wait_async();  // K (and first-iter Q) async transfers complete
    __syncthreads();

    // S^T tiles: acc t holds keys [16t,16t+16) x 16 queries (query = ln)
    v8f accS[4];
#pragma unroll
    for (int t = 0; t < 4; t++) accS[t] = (v8f){0, 0, 0, 0, 0, 0, 0, 0};
#pragma unroll
    for (int c = 0; c < 2; c++) {  // D=64 reduction in two 32-chunks
      Frag bQ;
      ld_frag(bQ, Qs, wave * 16 + ln, 72, c * 32, hl);  // column q of Q^T = row q of Q
#pragma unroll
      for (int t = 0; t < 4; t++) {
        Frag aK;
        ld_frag(aK, Ks, t * 16 + ln, 72, c * 32, hl);
        accS[t] = wmma_bf16(aK, bQ, accS[t]);
      }
    }

    // online softmax: this lane owns query ln, keys 16t + hl*8 + r
    float tmax = -1e30f;
#pragma unroll
    for (int t = 0; t < 4; t++)
#pragma unroll
      for (int r = 0; r < 8; r++) tmax = fmaxf(tmax, accS[t][r]);
    tmax = fmaxf(tmax, __shfl_xor(tmax, 16, 32));
    float m_new = fmaxf(m_run, tmax);
    float alpha = exp2f(m_run - m_new);
    float P[4][8];
    float tsum = 0.0f;
#pragma unroll
    for (int t = 0; t < 4; t++)
#pragma unroll
      for (int r = 0; r < 8; r++) {
        float pv = exp2f(accS[t][r] - m_new);
        P[t][r] = pv;
        tsum += pv;
      }
    tsum += __shfl_xor(tsum, 16, 32);
    l_run = l_run * alpha + tsum;
    m_run = m_new;

    // rescale O by per-query alpha (O rows are queries hl*8+r; stats at lane q)
#pragma unroll
    for (int r = 0; r < 8; r++) {
      float ar = __shfl(alpha, hl * 8 + r, 32);
#pragma unroll
      for (int dt = 0; dt < 4; dt++) accO[dt][r] *= ar;
    }

    // P (S^T C-layout) converts in-register to A-operand of P^T: no LDS trip
    Frag p0, p1;
#pragma unroll
    for (int r = 0; r < 8; r++) {
      p0.s[r]     = f2bf(P[0][r]);
      p0.s[8 + r] = f2bf(P[1][r]);
      p1.s[r]     = f2bf(P[2][r]);
      p1.s[8 + r] = f2bf(P[3][r]);
    }
    // O[16q x 64d] += P^T(16q x 64k) * V(64k x 64d)
#pragma unroll
    for (int dt = 0; dt < 4; dt++) {
      Frag bV0, bV1;
      ld_frag(bV0, Vt, dt * 16 + ln, 72, 0, hl);
      ld_frag(bV1, Vt, dt * 16 + ln, 72, 32, hl);
      accO[dt] = wmma_bf16(p0, bV0, accO[dt]);
      accO[dt] = wmma_bf16(p1, bV1, accO[dt]);
    }
  }

  // normalize and write [B,N,C] bf16 rows for the projection GEMM
#pragma unroll
  for (int r = 0; r < 8; r++) {
    float lr = __shfl(l_run, hl * 8 + r, 32);
    float linv = 1.0f / lr;
    int npos = qbase + wave * 16 + hl * 8 + r;
#pragma unroll
    for (int dt = 0; dt < 4; dt++) {
      int col = h * DDIM + dt * 16 + ln;
      Og[(size_t)(b * NPOS + npos) * CDIM + col] = f2bf(accO[dt][r] * linv);
    }
  }
}

// ---------------------------------------------------------------------------
extern "C" void kernel_launch(void* const* d_in, const int* in_sizes, int n_in,
                              void* d_out, int out_size, void* d_ws, size_t ws_size,
                              hipStream_t stream) {
  (void)in_sizes; (void)n_in; (void)out_size; (void)ws_size;
  const float* x      = (const float*)d_in[0];
  const float* w_qkv  = (const float*)d_in[1];
  const float* w_proj = (const float*)d_in[2];
  const float* b_proj = (const float*)d_in[3];
  float* out = (float*)d_out;

  char* p = (char*)d_ws;
  auto take = [&](size_t elems) {
    unsigned short* r = (unsigned short*)p;
    p += ((elems * 2 + 255) / 256) * 256;
    return r;
  };
  unsigned short* x_bf  = take((size_t)MROWS * CDIM);      // 8 MB
  unsigned short* wq_bf = take((size_t)CDIM * 3 * CDIM);   // 6 MB
  unsigned short* wp_bf = take((size_t)CDIM * CDIM);       // 2 MB
  unsigned short* q_bf  = take((size_t)MROWS * CDIM);      // [B,H,N,D] 8 MB
  unsigned short* k_bf  = take((size_t)MROWS * CDIM);
  unsigned short* v_bf  = take((size_t)MROWS * CDIM);
  unsigned short* oh_bf = take((size_t)MROWS * CDIM);      // attention out rows

  cast_f32_bf16<<<1024, 256, 0, stream>>>(x, x_bf, MROWS * CDIM);
  cast_f32_bf16<<<1024, 256, 0, stream>>>(w_qkv, wq_bf, CDIM * 3 * CDIM);
  cast_f32_bf16<<<512, 256, 0, stream>>>(w_proj, wp_bf, CDIM * CDIM);

  // QKV GEMM: [4096,1024] x [1024,3072] -> scatter q/k/v (q pre-scaled)
  gemm_wmma<1><<<dim3(3 * CDIM / 256, MROWS / 64), 256, 0, stream>>>(
      x_bf, wq_bf, 3 * CDIM, CDIM, nullptr, nullptr, q_bf, k_bf, v_bf);

  // Flash attention: grid (N/128 query blocks, B*H heads)
  attn_kernel<<<dim3(NPOS / 128, 2 * HDIM), 256, 0, stream>>>(q_bf, k_bf, v_bf, oh_bf);

  // Projection: [4096,1024] x [1024,1024] + bias -> f32 output
  gemm_wmma<0><<<dim3(CDIM / 256, MROWS / 64), 256, 0, stream>>>(
      oh_bf, wp_bf, CDIM, CDIM, out, b_proj, nullptr, nullptr, nullptr);
}
